// EdgeBlock_78872779424030
// MI455X (gfx1250) — matrix operand order repaired and checked by reading, not compiled, not run on hardware
//
#include <hip/hip_runtime.h>
#include <stdint.h>

#define N_NODES 50000
#define N_EDGES 500000
#define NODE_D 128
#define EDGE_D 128
#define IN_D   384   // EDGE_D + 2*NODE_D
#define OUT_D  128
#define KSTEPS 12    // IN_D / 32
#define NTILES 8     // OUT_D / 16
#define WBYTES (KSTEPS * NTILES * 512 * 2)  // 98304 B of packed bf16 W fragments

typedef __attribute__((ext_vector_type(16))) __bf16 v16bf;
typedef __attribute__((ext_vector_type(8)))  float  v8f;
typedef __attribute__((ext_vector_type(4)))  float  f4;

// ---------------------------------------------------------------------------
// Prep kernel: W [OUT_D x IN_D] f32 -> bf16, pre-swizzled into this file's contiguous-k WMMA
// B-fragment layout (32x16 bf16 per fragment):
//   fragment id = kstep*NTILES + ntile
//   per fragment: lane in [0,32), slot j in [0,16)
//     n = ntile*16 + (lane & 15)
//     k = kstep*32 + (lane >> 4)*16 + j   (lanes 0-15: K=0..15, 16-31: K=16..31)
//   stored flat at  frag*512 + lane*16 + j
// ---------------------------------------------------------------------------
__global__ void pack_W_bf16(const float* __restrict__ W, __bf16* __restrict__ wpack) {
  int tid   = blockIdx.x * blockDim.x + threadIdx.x;     // 0 .. 49151
  int frag  = tid >> 9;
  int r     = tid & 511;
  int lane  = r >> 4;
  int j     = r & 15;
  int kstep = frag >> 3;
  int ntile = frag & 7;
  int n = ntile * 16 + (lane & 15);
  int k = kstep * 32 + (lane >> 4) * 16 + j;
  wpack[tid] = (__bf16)W[n * IN_D + k];
}

// ---------------------------------------------------------------------------
// Main kernel: 8 waves / WG, 16 edges per wave (128 edges / WG).
//  - Packed W fragments staged into LDS via global_load_async_to_lds_b128
//    (ASYNCcnt path), then read back as ds_load_b128.  The LDS address
//    operand is derived from the ldsW pointer itself (low 32 bits of a
//    generic LDS pointer == hardware LDS offset, ISA 10.2) so the asm
//    visibly captures ldsW and the compiler keeps the ds_load reads.
//  - A fragments gathered from global (edge row / receiver row / sender row;
//    node table is L2-resident), software-pipelined one kstep ahead.
//  - B fragments double-buffered in registers across the 8 N-tiles.
// ---------------------------------------------------------------------------
__global__ __launch_bounds__(256) void edge_mlp_wmma(
    const float* __restrict__ nodes, const float* __restrict__ edges,
    const int* __restrict__ recv, const int* __restrict__ send,
    const __bf16* __restrict__ wpack, const float* __restrict__ bias,
    float* __restrict__ out)
{
  __shared__ __align__(16) char ldsW[WBYTES];   // 96 KB of 320 KB/WGP

  const int lane  = threadIdx.x & 31;
  const int wave  = threadIdx.x >> 5;
  const int strip = blockIdx.x * 128 + wave * 16;
  const int mrow  = lane & 15;
  const int khalf = lane >> 4;

  // ---- async bulk copy of packed W into LDS (96 KB, 16 B per lane-op) ----
  {
    unsigned t = threadIdx.x;
#pragma unroll
    for (int i = 0; i < WBYTES / (256 * 16); ++i) {      // 24 iterations
      unsigned byteoff = (i * 256 + t) * 16;
      // low 32 bits of generic LDS pointer == LDS byte address; this also
      // captures ldsW so the "memory"-clobber asm is known to write it.
      unsigned ldsoff = (unsigned)(uintptr_t)(ldsW + byteoff);
      asm volatile("global_load_async_to_lds_b128 %0, %1, %2 offset:0"
                   :: "v"(ldsoff), "v"(byteoff), "s"(wpack)
                   : "memory");
    }
    asm volatile("s_wait_asynccnt 0x0" ::: "memory");
  }
  __syncthreads();

  // ---- per-lane A row (fixed for whole kernel); clamp for tail tile ----
  int e  = strip + mrow;
  int ec = (e < N_EDGES) ? e : (N_EDGES - 1);
  const float* erow = edges + (long)ec * EDGE_D;
  const float* rrow = nodes + (long)recv[ec] * NODE_D;
  const float* srow = nodes + (long)send[ec] * NODE_D;

  // A source for a given kstep: each 32-wide K window lives in one source.
  auto srcp = [&](int ks) -> const float* {
    return (ks < 4) ? (erow + ks * 32)
         : (ks < 8) ? (rrow + ks * 32 - 128)
                    : (srow + ks * 32 - 256);
  };

  // ---- accumulators init with bias (col n = lane&15, same for all rows) ----
  v8f acc[NTILES];
#pragma unroll
  for (int nt = 0; nt < NTILES; ++nt) {
    float bv = bias[nt * 16 + mrow];
    acc[nt] = (v8f){bv, bv, bv, bv, bv, bv, bv, bv};
  }

  const v16bf* ldsB = (const v16bf*)ldsW;   // 32 v16bf per fragment

  // ---- prologue: issue A loads for kstep 0 ----
  f4 lo0, hi0, lo1, hi1;
  {
    const float* s0 = srcp(0) + khalf * 16;
    lo0 = *(const f4*)(s0);
    hi0 = *(const f4*)(s0 + 4);
    lo1 = *(const f4*)(s0 + 8);
    hi1 = *(const f4*)(s0 + 12);
  }

#pragma unroll
  for (int kstep = 0; kstep < KSTEPS; ++kstep) {
    // pack A fragment (contiguous k = khalf*16 + s, matching B) from staged f32 registers
    v16bf afrag;
#pragma unroll
    for (int t = 0; t < 4; ++t) {
      afrag[t]      = (__bf16)lo0[t];
      afrag[4 + t]  = (__bf16)hi0[t];
      afrag[8 + t]  = (__bf16)lo1[t];
      afrag[12 + t] = (__bf16)hi1[t];
    }

    // software pipeline: issue next kstep's A gather now
    if (kstep + 1 < KSTEPS) {
      const float* sn = srcp(kstep + 1) + khalf * 16;
      lo0 = *(const f4*)(sn);
      hi0 = *(const f4*)(sn + 4);
      lo1 = *(const f4*)(sn + 8);
      hi1 = *(const f4*)(sn + 12);
    }

    // 8 N-tiles with register double-buffered B fragments from LDS
    const v16bf* bk = ldsB + kstep * NTILES * 32;
    v16bf bcur = bk[lane];
#pragma unroll
    for (int nt = 0; nt < NTILES; ++nt) {
      v16bf bnxt = bcur;
      if (nt + 1 < NTILES) bnxt = bk[(nt + 1) * 32 + lane];
      acc[nt] = __builtin_amdgcn_wmma_f32_16x16x32_bf16(
          false, afrag, false, bcur, (short)0, acc[nt], false, false);
      bcur = bnxt;
    }
  }

  // ---- store: row m = r + 8*khalf, col n = nt*16 + (lane&15) ----
#pragma unroll
  for (int nt = 0; nt < NTILES; ++nt) {
#pragma unroll
    for (int r = 0; r < 8; ++r) {
      int ee = strip + r + 8 * khalf;
      if (ee < N_EDGES)
        out[(long)ee * OUT_D + nt * 16 + mrow] = acc[nt][r];
    }
  }
}

extern "C" void kernel_launch(void* const* d_in, const int* in_sizes, int n_in,
                              void* d_out, int out_size, void* d_ws, size_t ws_size,
                              hipStream_t stream) {
  const float* nodes = (const float*)d_in[0];
  const float* edges = (const float*)d_in[1];
  const int*   recv  = (const int*)d_in[2];
  const int*   send  = (const int*)d_in[3];
  const float* W     = (const float*)d_in[4];
  const float* b     = (const float*)d_in[5];
  float* out = (float*)d_out;
  __bf16* wpack = (__bf16*)d_ws;   // 96 KB of packed B fragments

  pack_W_bf16<<<(OUT_D * IN_D) / 256, 256, 0, stream>>>(W, wpack);

  int grid = (N_EDGES + 127) / 128;
  edge_mlp_wmma<<<grid, 256, 0, stream>>>(nodes, edges, recv, send, wpack, b, out);
}
